// ReadoutLayer_28613072126568
// MI455X (gfx1250) — matrix operand order repaired and checked by reading, not compiled, not run on hardware
//
#include <hip/hip_runtime.h>
#include <hip/hip_bf16.h>

// Problem constants (from reference): B=64, T=1000, I=512, H=512.
#define NB   64
#define NT   1000
#define NI   512
#define NH   512
#define TT   64            // timesteps per tile (4 WMMA M-tiles)
#define A_LO 0.8187307530779818f   // exp(-1/5)
#define A_HI 0.9607894391523232f   // exp(-1/25)

typedef __attribute__((ext_vector_type(16))) __bf16     v16bf;
typedef __attribute__((ext_vector_type(8)))  float      v8f;
typedef __attribute__((ext_vector_type(4)))  unsigned   v4u;

// ---- fp32 -> bf16 round-to-nearest-even, packed pair ----
__device__ __forceinline__ unsigned bf16rne(float f) {
    unsigned u = __float_as_uint(f);
    return (u + 0x7FFFu + ((u >> 16) & 1u)) >> 16;
}
__device__ __forceinline__ unsigned pk2(float lo, float hi) {
    return bf16rne(lo) | (bf16rne(hi) << 16);
}

// =====================================================================
// Kernel 1: pre-swizzle W (fp32 [H][I]) into bf16 B-fragment layout.
// Fragment (ks, nb) covers B[k0=32*ks .. +31][n0=16*nb .. +15], where
// B[k][n] = W[n][k].  Layout used here (contiguous k order, 16-bit B 32x16, wave32):
//   lane l: n = l&15, hi = l>>4; dword v (0..7) holds bf16 pair
//   k = 32*ks + 16*hi + 2*v (+1).
// Storage: dword index = ((ks*32 + nb)*32 + l)*8 + v  -> each lane's 8
// dwords are 32 contiguous bytes => two global_load_b128 per fragment.
// Total: 16*32*32*8 dwords = 512 KB in d_ws.
// =====================================================================
__global__ void wswz_kernel(const float* __restrict__ W, unsigned* __restrict__ ws) {
    int g  = blockIdx.x * 256 + threadIdx.x;   // 0..131071
    int v  = g & 7;
    int l  = (g >> 3) & 31;
    int nb = (g >> 8) & 31;
    int ks = g >> 13;
    int n  = nb * 16 + (l & 15);
    int k  = ks * 32 + 16 * (l >> 4) + 2 * v;
    float f0 = W[n * NI + k];
    float f1 = W[n * NI + k + 1];
    ws[g] = pk2(f0, f1);
}

// =====================================================================
// Kernel 2: fused GEMM (bf16 WMMA) + leaky-integrate + softmax-accumulate.
// One block per batch element b; 512 threads = 16 waves.
// LDS: A-staging (bf16, 64 rows x 1040B stride = 65 KB) overlapped with
//      wx/ut/e buffer (f32 [64][512] = 128 KB) + invsum[64].
// =====================================================================
#define AROWSTRIDE 1040          // 1024 + 16B pad -> conflict-free ds_load_b128
#define WXBYTES    (TT * NH * 4) // 131072

__global__ __launch_bounds__(512, 1)
void readout_main(const float* __restrict__ x,
                  const unsigned* __restrict__ wswz,
                  const float* __restrict__ alpha,
                  const float* __restrict__ u0,
                  float* __restrict__ out) {
    __shared__ __align__(16) unsigned char smem[WXBYTES + 256];
    float*          wxf    = (float*)smem;              // [TT][NH] f32
    float*          invsum = (float*)(smem + WXBYTES);  // [TT]
    unsigned char*  aBase  = smem;                      // bf16 A staging (overlapped)

    const int tid = threadIdx.x;
    const int b   = blockIdx.x;
    const int w   = tid >> 5;        // wave 0..15
    const int l   = tid & 31;        // lane
    const int lm  = l & 15;
    const int hi  = l >> 4;

    const float a  = fminf(fmaxf(alpha[tid], A_LO), A_HI);
    const float ca = 1.0f - a;
    float u = u0[b * NH + tid];
    float o = 0.0f;

    const float*  xb  = x + (size_t)b * NT * NI;
    const v16bf*  wsb = (const v16bf*)wswz;   // 32B-stride fragment lanes

    for (int tile = 0; tile < (NT + TT - 1) / TT; ++tile) {
        const int t0  = tile * TT;
        const int ttv = (NT - t0 < TT) ? (NT - t0) : TT;

        __syncthreads();   // previous tile's phase-3 reads of wxf are done

        // ---- stage A tile: x[t0..t0+63][0..511] -> bf16 LDS (zero-pad tail)
        #pragma unroll
        for (int it = 0; it < 16; ++it) {
            int s   = tid + it * 512;          // float4 slot, 8192 total
            int m   = s >> 7;                  // row (128 float4 per row)
            int col = (s & 127) << 2;
            float4 f;
            if (t0 + m < NT) f = *(const float4*)(xb + (size_t)(t0 + m) * NI + col);
            else             f = make_float4(0.f, 0.f, 0.f, 0.f);
            uint2 p;
            p.x = pk2(f.x, f.y);
            p.y = pk2(f.z, f.w);
            *(uint2*)(aBase + m * AROWSTRIDE + col * 2) = p;
        }
        __syncthreads();

        // ---- GEMM: wave w computes M-tiles 0..3, N-tiles {2w, 2w+1}
        v8f acc[4][2];
        #pragma unroll
        for (int mt = 0; mt < 4; ++mt)
            #pragma unroll
            for (int nt = 0; nt < 2; ++nt)
                acc[mt][nt] = (v8f){0.f,0.f,0.f,0.f,0.f,0.f,0.f,0.f};

        const unsigned char* aRow[4];
        #pragma unroll
        for (int mt = 0; mt < 4; ++mt)
            aRow[mt] = aBase + (mt * 16 + lm) * AROWSTRIDE + hi * 32;

        #pragma unroll 4
        for (int ks = 0; ks < 16; ++ks) {
            v16bf bfr[2];
            #pragma unroll
            for (int nt = 0; nt < 2; ++nt)
                bfr[nt] = wsb[(ks * 32 + (2 * w + nt)) * 32 + l];
            #pragma unroll
            for (int mt = 0; mt < 4; ++mt) {
                union { v4u q[2]; v16bf v; } af;
                af.q[0] = *(const v4u*)(aRow[mt] + ks * 64);        // k = 16*hi + 0..7
                af.q[1] = *(const v4u*)(aRow[mt] + ks * 64 + 16);   // k = 16*hi + 8..15
                #pragma unroll
                for (int nt = 0; nt < 2; ++nt)
                    acc[mt][nt] = __builtin_amdgcn_wmma_f32_16x16x32_bf16(
                        false, af.v, false, bfr[nt], (short)0, acc[mt][nt], false, false);
            }
        }
        __syncthreads();   // all waves finished reading A region

        // ---- write accumulators to wx (C layout: reg r -> M=r+8*hi, N=lm)
        #pragma unroll
        for (int mt = 0; mt < 4; ++mt)
            #pragma unroll
            for (int nt = 0; nt < 2; ++nt)
                #pragma unroll
                for (int r = 0; r < 8; ++r)
                    wxf[(mt * 16 + r + 8 * hi) * NH + ((2 * w + nt) * 16 + lm)] = acc[mt][nt][r];
        __syncthreads();

        // ---- phase 1: sequential leaky integration (thread owns h = tid)
        for (int tt = 0; tt < ttv; ++tt) {
            float v = wxf[tt * NH + tid];
            u = a * u + ca * v;
            wxf[tt * NH + tid] = u;          // store u_t in place
        }
        __syncthreads();

        // ---- phase 2: per-row softmax prep (wave w handles rows 4w..4w+3)
        #pragma unroll
        for (int i = 0; i < 4; ++i) {
            int tt = w * 4 + i;
            if (tt < ttv) {
                float mx = -3.402823466e38f;
                #pragma unroll
                for (int q = 0; q < 16; ++q)
                    mx = fmaxf(mx, wxf[tt * NH + l + 32 * q]);
                #pragma unroll
                for (int off = 16; off; off >>= 1)
                    mx = fmaxf(mx, __shfl_xor(mx, off, 32));
                float sm = 0.f;
                #pragma unroll
                for (int q = 0; q < 16; ++q) {
                    int idx = tt * NH + l + 32 * q;
                    float e = __expf(wxf[idx] - mx);
                    wxf[idx] = e;            // store exp in place
                    sm += e;
                }
                #pragma unroll
                for (int off = 16; off; off >>= 1)
                    sm += __shfl_xor(sm, off, 32);
                if (l == 0) invsum[tt] = 1.0f / sm;
            }
        }
        __syncthreads();

        // ---- phase 3: accumulate normalized softmax into out
        for (int tt = 0; tt < ttv; ++tt)
            o = fmaf(wxf[tt * NH + tid], invsum[tt], o);
    }

    out[b * NH + tid] = o;
}

// =====================================================================
// Inputs (setup_inputs order): d_in[0]=x f32[64*1000*512], d_in[1]=W f32[512*512],
// d_in[2]=alpha f32[512], d_in[3]=u0 f32[64*512]. Output: f32[64*512].
// Workspace: needs 512 KB for pre-swizzled bf16 W fragments.
// =====================================================================
extern "C" void kernel_launch(void* const* d_in, const int* in_sizes, int n_in,
                              void* d_out, int out_size, void* d_ws, size_t ws_size,
                              hipStream_t stream) {
    const float* x     = (const float*)d_in[0];
    const float* W     = (const float*)d_in[1];
    const float* alpha = (const float*)d_in[2];
    const float* u0    = (const float*)d_in[3];
    float*       outp  = (float*)d_out;
    unsigned*    wsw   = (unsigned*)d_ws;     // 131072 dwords = 512 KB

    wswz_kernel<<<512, 256, 0, stream>>>(W, wsw);
    readout_main<<<NB, 512, 0, stream>>>(x, wsw, alpha, u0, outp);
}
